// GraphSAGEGraphLevel_48670569398702
// MI455X (gfx1250) — matrix operand order repaired and checked by reading, not compiled, not run on hardware
//
#include <hip/hip_runtime.h>
#include <hip/hip_bf16.h>

typedef __bf16 bf16_t;
typedef __attribute__((ext_vector_type(16))) __bf16 v16bf;
typedef __attribute__((ext_vector_type(8)))  __bf16 v8bf;
typedef __attribute__((ext_vector_type(8)))  float  v8f;

#define HDIM 128

// ---------------- feature assembly: h0 = [x | xdims | emb[xsttype] | zero-pad] in bf16 ----------------
__global__ __launch_bounds__(256) void k_assemble(
    const float* __restrict__ x, const float* __restrict__ xdims,
    const int* __restrict__ xsttype, const float* __restrict__ emb,
    bf16_t* __restrict__ h0, int N)
{
  long long t = (long long)blockIdx.x * blockDim.x + threadIdx.x;
  if (t >= (long long)N * HDIM) return;
  int node = (int)(t >> 7);
  int c    = (int)(t & 127);
  float v = 0.0f;
  if (c < 60)      v = x[node * 60 + c];
  else if (c < 62) v = xdims[node * 2 + (c - 60)];
  else if (c < 74) v = emb[xsttype[node] * 12 + (c - 62)];
  h0[t] = (bf16_t)v;
}

// ---------------- degree / inverse degree / per-graph node count ----------------
__global__ __launch_bounds__(256) void k_degree(const int* __restrict__ dst,
                                                float* __restrict__ deg, int E)
{
  int t = blockIdx.x * blockDim.x + threadIdx.x;
  if (t < E) atomicAdd(&deg[dst[t]], 1.0f);
}

__global__ __launch_bounds__(256) void k_invdeg(const float* __restrict__ deg,
                                                float* __restrict__ ivd, int N)
{
  int t = blockIdx.x * blockDim.x + threadIdx.x;
  if (t < N) ivd[t] = 1.0f / fmaxf(deg[t], 1.0f);
}

__global__ __launch_bounds__(256) void k_count(const int* __restrict__ batch,
                                               float* __restrict__ cnt, int N)
{
  int t = blockIdx.x * blockDim.x + threadIdx.x;
  if (t < N) atomicAdd(&cnt[batch[t]], 1.0f);
}

// ---------------- edge aggregation: agg[dst] += h[src] (fp32 accumulation) ----------------
__global__ __launch_bounds__(256) void k_aggregate(
    const bf16_t* __restrict__ hin, const int* __restrict__ src,
    const int* __restrict__ dst, float* __restrict__ agg, int E)
{
  long long t = (long long)blockIdx.x * blockDim.x + threadIdx.x;
  if (t >= (long long)E * HDIM) return;
  int e = (int)(t >> 7);
  int f = (int)(t & 127);
  float v = (float)hin[(long long)src[e] * HDIM + f];
  atomicAdd(&agg[(long long)dst[e] * HDIM + f], v);
}

// ---------------- pack weight [in_dim,128] fp32 -> WMMA B-fragment bf16 layout ----------------
// fragment f = kt*8+nt ; lane L holds column n = nt*16+(L&15), K = kt*32 + 8*(L>>4) + (i&7) + 16*(i>>3) (i=0..15)
__global__ __launch_bounds__(256) void k_pack_w(const float* __restrict__ W,
                                                bf16_t* __restrict__ P, int in_dim)
{
  int p = blockIdx.x * 256 + threadIdx.x;   // 0 .. 16383
  int i  = p & 15;
  int L  = (p >> 4) & 31;
  int nt = (p >> 9) & 7;
  int kt = p >> 12;
  int k = kt * 32 + ((L >> 4) << 3) + (i & 7) + ((i >> 3) << 4);
  int n = nt * 16 + (L & 15);
  float v = (k < in_dim) ? W[k * HDIM + n] : 0.0f;
  P[p] = (bf16_t)v;
}

// ---------------- build 16x32 bf16 A fragment from an LDS tile (ISA layout) ----------------
// lane L: row m = L&15 ; K = kbase + {0..7} and kbase+16 + {0..7}, kbase = kt*32 + 8*(L>>4)
__device__ inline v16bf make_a_frag(const bf16_t* tile, int m, int kbase)
{
  v8bf lo = *(const v8bf*)(tile + m * HDIM + kbase);
  v8bf hi = *(const v8bf*)(tile + m * HDIM + kbase + 16);
  v16bf a;
#pragma unroll
  for (int i = 0; i < 8; ++i) { a[i] = lo[i]; a[i + 8] = hi[i]; }
  return a;
}

// ---------------- SAGE layer: out = relu(LN(agg*invdeg @ Wl + bl + h @ Wr)) ----------------
// block = 16 rows x 128 cols; 8 waves (wave32), each owns one 16x16 n-tile.
__global__ __launch_bounds__(256) void k_sage_layer(
    const bf16_t* __restrict__ hin, const float* __restrict__ agg,
    const float* __restrict__ invdeg,
    const bf16_t* __restrict__ wlp, const bf16_t* __restrict__ wrp,
    const float* __restrict__ bl, const float* __restrict__ gamma,
    const float* __restrict__ beta, bf16_t* __restrict__ hout, int N)
{
  __shared__ bf16_t sH[16 * HDIM];   // root features (bf16)
  __shared__ bf16_t sA[16 * HDIM];   // mean-aggregated neighbors (bf16, pre-scaled)
  __shared__ float  sO[16 * HDIM];   // f32 output tile
  __shared__ float  sMu[16], sRs[16];

  const int tid  = threadIdx.x;
  const int row0 = blockIdx.x * 16;

  { // cooperative tile load: 8 elements / thread
    int r  = tid >> 4;
    int c0 = (tid & 15) * 8;
    int rr = min(row0 + r, N - 1);
    *(v8bf*)(sH + r * HDIM + c0) = *(const v8bf*)(hin + (long long)rr * HDIM + c0);
    float s = invdeg[rr];
    const float* ap = agg + (long long)rr * HDIM + c0;
    bf16_t* dp = sA + r * HDIM + c0;
#pragma unroll
    for (int j = 0; j < 8; ++j) dp[j] = (bf16_t)(ap[j] * s);
  }
  __syncthreads();

  const int lane  = tid & 31;
  const int nt    = tid >> 5;          // n-tile per wave
  const int m     = lane & 15;
  const int khalf = (lane >> 4) * 8;

  v8f acc = {};
#pragma unroll
  for (int kt = 0; kt < 4; ++kt) {
    const int kbase = kt * 32 + khalf;
    const int fbase = ((kt * 8 + nt) * 32 + lane) * 16;
    v16bf aA = make_a_frag(sA, m, kbase);
    v16bf bL = *(const v16bf*)(wlp + fbase);
    acc = __builtin_amdgcn_wmma_f32_16x16x32_bf16(false, aA, false, bL,
                                                  (short)0, acc, false, false);
    v16bf aH = make_a_frag(sH, m, kbase);
    v16bf bR = *(const v16bf*)(wrp + fbase);
    acc = __builtin_amdgcn_wmma_f32_16x16x32_bf16(false, aH, false, bR,
                                                  (short)0, acc, false, false);
  }

  // C/D layout: lane 0-15 -> (M=r, N=lane); lane 16-31 -> (M=8+r, N=lane-16)
  const int n     = nt * 16 + (lane & 15);
  const int mbase = (lane >> 4) * 8;
  const float bv  = bl[n];
#pragma unroll
  for (int r = 0; r < 8; ++r) sO[(mbase + r) * HDIM + n] = acc[r] + bv;
  __syncthreads();

  if (tid < 16) {  // per-row LayerNorm statistics
    float mu = 0.0f;
    for (int k2 = 0; k2 < HDIM; ++k2) mu += sO[tid * HDIM + k2];
    mu *= (1.0f / HDIM);
    float var = 0.0f;
    for (int k2 = 0; k2 < HDIM; ++k2) { float d = sO[tid * HDIM + k2] - mu; var += d * d; }
    var *= (1.0f / HDIM);
    sMu[tid] = mu;
    sRs[tid] = rsqrtf(var + 1e-5f);
  }
  __syncthreads();

  { // normalize + relu + store bf16
    int r  = tid >> 4;
    int c0 = (tid & 15) * 8;
    if (row0 + r < N) {
      float mu = sMu[r], rs = sRs[r];
#pragma unroll
      for (int j = 0; j < 8; ++j) {
        int cc = c0 + j;
        float v = (sO[r * HDIM + cc] - mu) * rs * gamma[cc] + beta[cc];
        hout[(long long)(row0 + r) * HDIM + cc] = (bf16_t)fmaxf(v, 0.0f);
      }
    }
  }
}

// ---------------- graph pooling: mean (atomicAdd) + max (u32 atomicMax on non-neg floats) ----------------
__global__ __launch_bounds__(256) void k_pool(
    const bf16_t* __restrict__ h, const int* __restrict__ batch,
    float* __restrict__ zsum, unsigned int* __restrict__ zmax, int N)
{
  long long t = (long long)blockIdx.x * blockDim.x + threadIdx.x;
  if (t >= (long long)N * HDIM) return;
  int node = (int)(t >> 7);
  int f    = (int)(t & 127);
  float v  = (float)h[t];
  int g    = batch[node];
  atomicAdd(&zsum[g * HDIM + f], v);
  atomicMax(&zmax[g * HDIM + f], __float_as_uint(v));   // v >= 0 post-ReLU
}

// ---------------- tiny MLP head: [mean|max](256) -> 50 -> 50 -> 10, ReLU ----------------
__global__ __launch_bounds__(64) void k_mlp(
    const float* __restrict__ zsum, const unsigned int* __restrict__ zmax,
    const float* __restrict__ cnt,
    const float* __restrict__ Wf0, const float* __restrict__ bf0,
    const float* __restrict__ Wf1, const float* __restrict__ bf1,
    const float* __restrict__ Wf2, const float* __restrict__ bf2,
    float* __restrict__ out)
{
  __shared__ float z[2 * HDIM];
  __shared__ float t0[50], t1[50];
  int g = blockIdx.x, tid = threadIdx.x;
  float c = fmaxf(cnt[g], 1.0f);
  for (int i = tid; i < HDIM; i += 64) {
    z[i]        = zsum[g * HDIM + i] / c;
    z[HDIM + i] = __uint_as_float(zmax[g * HDIM + i]);
  }
  __syncthreads();
  if (tid < 50) {
    float s = bf0[tid];
    for (int k = 0; k < 2 * HDIM; ++k) s += z[k] * Wf0[k * 50 + tid];
    t0[tid] = fmaxf(s, 0.0f);
  }
  __syncthreads();
  if (tid < 50) {
    float s = bf1[tid];
    for (int k = 0; k < 50; ++k) s += t0[k] * Wf1[k * 50 + tid];
    t1[tid] = fmaxf(s, 0.0f);
  }
  __syncthreads();
  if (tid < 10) {
    float s = bf2[tid];
    for (int k = 0; k < 50; ++k) s += t1[k] * Wf2[k * 10 + tid];
    out[g * 10 + tid] = fmaxf(s, 0.0f);
  }
}

// ---------------- host-side orchestration ----------------
extern "C" void kernel_launch(void* const* d_in, const int* in_sizes, int n_in,
                              void* d_out, int out_size, void* d_ws, size_t ws_size,
                              hipStream_t stream)
{
  const float* x     = (const float*)d_in[0];
  const float* xdims = (const float*)d_in[1];
  const int*   xst   = (const int*)d_in[2];
  const int*   ei    = (const int*)d_in[3];
  const int*   batch = (const int*)d_in[4];
  const float* emb   = (const float*)d_in[5];
  const float* Wl0 = (const float*)d_in[6];
  const float* bl0 = (const float*)d_in[7];
  const float* Wr0 = (const float*)d_in[8];
  const float* g0  = (const float*)d_in[9];
  const float* bn0 = (const float*)d_in[10];
  const float* Wl1 = (const float*)d_in[11];
  const float* bl1 = (const float*)d_in[12];
  const float* Wr1 = (const float*)d_in[13];
  const float* g1  = (const float*)d_in[14];
  const float* bn1 = (const float*)d_in[15];
  const float* Wf0 = (const float*)d_in[16];
  const float* bf0 = (const float*)d_in[17];
  const float* Wf1 = (const float*)d_in[18];
  const float* bf1 = (const float*)d_in[19];
  const float* Wf2 = (const float*)d_in[20];
  const float* bf2 = (const float*)d_in[21];

  const int N = in_sizes[2];
  const int E = in_sizes[3] / 2;
  const int G = out_size / 10;
  const int in_dim0 = in_sizes[6] / HDIM;   // 74
  const int* src = ei;
  const int* dst = ei + E;

  // carve scratch out of d_ws (256B aligned)
  char* p = (char*)d_ws;
  auto carve = [&](size_t bytes) { char* r = p; p += (bytes + 255) & ~(size_t)255; return r; };
  bf16_t* h0   = (bf16_t*)carve((size_t)N * HDIM * sizeof(bf16_t));
  bf16_t* h1   = (bf16_t*)carve((size_t)N * HDIM * sizeof(bf16_t));
  float*  agg  = (float*) carve((size_t)N * HDIM * sizeof(float));
  float*  deg  = (float*) carve((size_t)N * sizeof(float));
  float*  ivd  = (float*) carve((size_t)N * sizeof(float));
  bf16_t* wl0p = (bf16_t*)carve(16384 * sizeof(bf16_t));
  bf16_t* wr0p = (bf16_t*)carve(16384 * sizeof(bf16_t));
  bf16_t* wl1p = (bf16_t*)carve(16384 * sizeof(bf16_t));
  bf16_t* wr1p = (bf16_t*)carve(16384 * sizeof(bf16_t));
  float*        zsum = (float*)       carve((size_t)G * HDIM * sizeof(float));
  unsigned int* zmax = (unsigned int*)carve((size_t)G * HDIM * sizeof(unsigned int));
  float*        cnt  = (float*)       carve((size_t)G * sizeof(float));

  hipMemsetAsync(deg,  0, (size_t)N * sizeof(float), stream);
  hipMemsetAsync(agg,  0, (size_t)N * HDIM * sizeof(float), stream);
  hipMemsetAsync(zsum, 0, (size_t)G * HDIM * sizeof(float), stream);
  hipMemsetAsync(zmax, 0, (size_t)G * HDIM * sizeof(unsigned int), stream);
  hipMemsetAsync(cnt,  0, (size_t)G * sizeof(float), stream);

  k_pack_w<<<64, 256, 0, stream>>>(Wl0, wl0p, in_dim0);
  k_pack_w<<<64, 256, 0, stream>>>(Wr0, wr0p, in_dim0);
  k_pack_w<<<64, 256, 0, stream>>>(Wl1, wl1p, HDIM);
  k_pack_w<<<64, 256, 0, stream>>>(Wr1, wr1p, HDIM);

  long long nh = (long long)N * HDIM;
  int blk_nh = (int)((nh + 255) / 256);
  k_assemble<<<blk_nh, 256, 0, stream>>>(x, xdims, xst, emb, h0, N);
  k_degree<<<(E + 255) / 256, 256, 0, stream>>>(dst, deg, E);
  k_invdeg<<<(N + 255) / 256, 256, 0, stream>>>(deg, ivd, N);
  k_count<<<(N + 255) / 256, 256, 0, stream>>>(batch, cnt, N);

  long long ne = (long long)E * HDIM;
  int blk_ne = (int)((ne + 255) / 256);
  int blk_rows = (N + 15) / 16;

  // layer 0
  k_aggregate<<<blk_ne, 256, 0, stream>>>(h0, src, dst, agg, E);
  k_sage_layer<<<blk_rows, 256, 0, stream>>>(h0, agg, ivd, wl0p, wr0p, bl0, g0, bn0, h1, N);
  // layer 1
  hipMemsetAsync(agg, 0, (size_t)N * HDIM * sizeof(float), stream);
  k_aggregate<<<blk_ne, 256, 0, stream>>>(h1, src, dst, agg, E);
  k_sage_layer<<<blk_rows, 256, 0, stream>>>(h1, agg, ivd, wl1p, wr1p, bl1, g1, bn1, h0, N);

  // pooling + head
  k_pool<<<blk_nh, 256, 0, stream>>>(h0, batch, zsum, zmax, N);
  k_mlp<<<G, 64, 0, stream>>>(zsum, zmax, cnt, Wf0, bf0, Wf1, bf1, Wf2, bf2, (float*)d_out);
}
